// OrbitalNetwork_44452911513676
// MI455X (gfx1250) — matrix operand order repaired and checked by reading, not compiled, not run on hardware
//
#include <hip/hip_runtime.h>
#include <math.h>

typedef _Float16 half_t;
typedef half_t v16h __attribute__((ext_vector_type(16)));
typedef half_t v8h  __attribute__((ext_vector_type(8)));
typedef float  v8f  __attribute__((ext_vector_type(8)));

#define N_NODES 50000
#define N_EDGES 800000
#define FDIM    16
#define ADIM    8
#define OUTD    16
#define NBASIS  10
#define HDIM    100
#define SHD     9
#define GDIM    8

#define H_PAD   112   // 7*16 columns for hidden
#define K1_PAD  32    // radial basis K padded to one WMMA K
#define K2_PAD  128   // hidden K padded to 4 WMMA K-chunks
#define N2      144   // F*SH = 16*9

#define WAVES   8
#define EPB     (WAVES * 16)   // edges per block

static __device__ inline v8f wmma16(v16h a, v16h b, v8f c) {
  // D = A(16x32 f16) * B(32x16 f16) + C(16x16 f32)
  return __builtin_amdgcn_wmma_f32_16x16x32_f16(
      false, a, false, b, (short)0, c, /*reuse_a=*/false, /*reuse_b=*/false);
}
static __device__ inline v8f wmma16_ra(v16h a, v16h b, v8f c) {
  // same, with reuse-A hint: previous instruction in the chain has identical A
  return __builtin_amdgcn_wmma_f32_16x16x32_f16(
      false, a, false, b, (short)0, c, /*reuse_a=*/true, /*reuse_b=*/false);
}

// Wave-local LDS ordering: all our LDS tiles are private to one wave, and
// CDNA5 LDS ops from the same wave complete in order, so a full workgroup
// barrier is unnecessary. s_wait_dscnt 0 + compiler memory barrier suffices.
static __device__ inline void wave_lds_fence() {
  asm volatile("s_wait_dscnt 0x0" ::: "memory");
}

// A-fragment (16x32 f16) from an LDS tile stored row-major [m][k], stride 32.
// Layout: lanes 0-15: halves 0..15 = K0..15;
//         lanes 16-31: halves 0..15 = K16..31.
static __device__ inline v16h load_A_lds(const half_t* s, int lane) {
  int m  = lane & 15;
  int kb = (lane & 16) ? 16 : 0;
  v8h lo = *(const v8h*)(s + m * 32 + kb);
  v8h hi = *(const v8h*)(s + m * 32 + kb + 8);
  v16h a;
#pragma unroll
  for (int i = 0; i < 8; ++i) { a[i] = lo[i]; a[8 + i] = hi[i]; }
  return a;
}

// B-fragment (32x16 f16) from global weights stored [n][k] (k contiguous, stride KS).
// Layout: lane gives N; halves give K (lanes 16-31 offset by +16).
static __device__ inline v16h load_B_glb(const half_t* g, int KS, int kbase, int n0, int lane) {
  int n  = n0 + (lane & 15);
  int kb = kbase + ((lane & 16) ? 16 : 0);
  return *(const v16h*)(g + n * KS + kb);
}

// ---------------------------------------------------------------- prep weights
__global__ __launch_bounds__(256) void k_prep(const float* fc_w1, const float* fc_w2,
                                              half_t* fc1h, half_t* fc2h) {
  int idx = blockIdx.x * blockDim.x + threadIdx.x;
  int tot1 = 2 * H_PAD * K1_PAD;           // [l][n][k]
  int tot2 = 2 * N2 * K2_PAD;              // [l][n][k]
  if (idx < tot1) {
    int k = idx % K1_PAD, n = (idx / K1_PAD) % H_PAD, l = idx / (K1_PAD * H_PAD);
    float v = (k < NBASIS && n < HDIM) ? fc_w1[l * NBASIS * HDIM + k * HDIM + n] : 0.0f;
    fc1h[idx] = (half_t)v;
  }
  if (idx < tot2) {
    int k = idx % K2_PAD, n = (idx / K2_PAD) % N2, l = idx / (K2_PAD * N2);
    float v = (k < HDIM) ? fc_w2[l * HDIM * N2 + k * N2 + n] : 0.0f;
    fc2h[idx] = (half_t)v;
  }
}

// ---------------------------------------------------------------- h = x@w_in + z@w_attr
__global__ __launch_bounds__(256) void k_node_init(const float* x, const float* z,
                                                   const float* w_in, const float* w_attr,
                                                   float* h) {
  int idx = blockIdx.x * blockDim.x + threadIdx.x;
  if (idx >= N_NODES * FDIM) return;
  int n = idx >> 4, f = idx & 15;
  float acc = 0.0f;
#pragma unroll
  for (int i = 0; i < FDIM; ++i) acc += x[n * FDIM + i] * w_in[i * FDIM + f];
#pragma unroll
  for (int a = 0; a < ADIM; ++a) acc += z[n * ADIM + a] * w_attr[a * FDIM + f];
  h[idx] = acc;
}

// ---------------------------------------------------------------- per-edge message pass
__global__ __launch_bounds__(256) void k_msg(const float* __restrict__ pos,
                                             const float* __restrict__ h,
                                             float* __restrict__ agg,
                                             const half_t* __restrict__ fc1h,
                                             const half_t* __restrict__ fc2h,
                                             const int* __restrict__ esrc,
                                             const int* __restrict__ edst,
                                             int layer) {
  __shared__ half_t s_emb[WAVES][16 * 32];     // radial basis A-tiles (K padded to 32)
  __shared__ float  s_sh [WAVES][16 * SHD];    // spherical harmonics
  __shared__ half_t s_hid[WAVES][16 * 32];     // staged hidden A-tile (one K-chunk)
  __shared__ half_t s_w  [WAVES][16 * N2];     // 16x144 weight tile (f16)

  const int lane = threadIdx.x & 31;
  const int wv   = threadIdx.x >> 5;
  const int eBase = blockIdx.x * EPB + wv * 16;   // E divisible by 128 -> always full

  // ---- geometry: lanes 0-15, one edge each
  if (lane < 16) {
    int e  = eBase + lane;
    int si = esrc[e], di = edst[e];
    float ex = pos[si * 3 + 0] - pos[di * 3 + 0];
    float ey = pos[si * 3 + 1] - pos[di * 3 + 1];
    float ez = pos[si * 3 + 2] - pos[di * 3 + 2];
    float r  = sqrtf(ex * ex + ey * ey + ez * ez);
    float iv = 1.0f / (r + 1e-12f);
    float ux = ex * iv, uy = ey * iv, uz = ez * iv;
    const float s3 = 1.7320508075688772f, s15 = 3.872983346207417f, s5 = 2.23606797749979f;
    float* sh = &s_sh[wv][lane * SHD];
    sh[0] = 1.0f;
    sh[1] = s3 * ux;  sh[2] = s3 * uy;  sh[3] = s3 * uz;
    sh[4] = s15 * ux * uy;
    sh[5] = s15 * uy * uz;
    sh[6] = 0.5f * s5 * (3.0f * uz * uz - 1.0f);
    sh[7] = s15 * uz * ux;
    sh[8] = 0.5f * s15 * (ux * ux - uy * uy);
    const float step = 3.5f / 11.0f;
    const float sq10 = 3.1622776601683795f;
#pragma unroll
    for (int k = 0; k < 32; ++k) {
      float v = 0.0f;
      if (k < NBASIS) {
        float center = 3.5f * (float)(k + 1) / 11.0f;
        float d = (r - center) / step;
        if (d > -1.0f && d < 1.0f) v = __cosf(0.5f * (float)M_PI * d) * sq10;
      }
      s_emb[wv][lane * 32 + k] = (half_t)v;
    }
  }
  wave_lds_fence();

  const v16h embA = load_A_lds(s_emb[wv], lane);
  const half_t* w1 = fc1h + layer * H_PAD * K1_PAD;
  const half_t* w2 = fc2h + layer * N2 * K2_PAD;

  v8f acc[9];
#pragma unroll
  for (int t = 0; t < 9; ++t) acc[t] = (v8f)(0.0f);

  const int mrow = (lane & 16) ? 8 : 0;
  const int ncol = lane & 15;

  for (int kc = 0; kc < 4; ++kc) {
    // matmul1: two 16-column hidden chunks -> SiLU -> stage as next A-tile
#pragma unroll
    for (int cc = 0; cc < 2; ++cc) {
      int c = 2 * kc + cc;                 // hidden column chunk, 0..7 (7 is pure pad)
      v8f hid = (v8f)(0.0f);
      if (c < 7) {
        v16h b1 = load_B_glb(w1, K1_PAD, 0, c * 16, lane);
        hid = wmma16(embA, b1, hid);
      }
#pragma unroll
      for (int v = 0; v < 8; ++v) {
        float xv = hid[v];
        // fast SiLU: x * rcp(1 + exp(-x))  (v_exp_f32 + v_rcp_f32, no IEEE div)
        float sv = (c < 7) ? (xv * __builtin_amdgcn_rcpf(1.0f + __expf(-xv))) : 0.0f;
        s_hid[wv][(v + mrow) * 32 + cc * 16 + ncol] = (half_t)sv;  // D layout -> row-major
      }
    }
    wave_lds_fence();

    // matmul2: accumulate 16x144 tile over this K-chunk (chain shares A -> reuse hint)
    v16h hA = load_A_lds(s_hid[wv], lane);
    {
      v16h b2 = load_B_glb(w2, K2_PAD, kc * 32, 0, lane);
      acc[0] = wmma16(hA, b2, acc[0]);
    }
#pragma unroll
    for (int t = 1; t < 9; ++t) {
      v16h b2 = load_B_glb(w2, K2_PAD, kc * 32, t * 16, lane);
      acc[t] = wmma16_ra(hA, b2, acc[t]);
    }
    wave_lds_fence();
  }

  // ---- spill W tile to LDS (f16)
#pragma unroll
  for (int t = 0; t < 9; ++t)
#pragma unroll
    for (int v = 0; v < 8; ++v)
      s_w[wv][(v + mrow) * N2 + t * 16 + ncol] = (half_t)acc[t][v];
  wave_lds_fence();

  // ---- msg[e,f] = h[src,f] * sum_k W[e, f*9+k] * sh[e,k]; scatter-add to agg[dst]
#pragma unroll
  for (int it = 0; it < 8; ++it) {
    int idx = it * 32 + lane;
    int el = idx >> 4, f = idx & 15;
    int e  = eBase + el;
    float dot = 0.0f;
#pragma unroll
    for (int k = 0; k < SHD; ++k)
      dot += (float)s_w[wv][el * N2 + f * SHD + k] * s_sh[wv][el * SHD + k];
    int si = esrc[e], di = edst[e];
    float m = h[si * FDIM + f] * dot;
    atomicAdd(&agg[di * FDIM + f], m);
  }
}

// ---------------------------------------------------------------- h += (agg/4) @ lin[l]
__global__ __launch_bounds__(256) void k_update(float* h, const float* agg,
                                                const float* lin, int layer) {
  int idx = blockIdx.x * blockDim.x + threadIdx.x;
  if (idx >= N_NODES * FDIM) return;
  int n = idx >> 4, f = idx & 15;
  const float inv_sqrt_nei = 0.25f;   // 1/sqrt(16)
  float acc = 0.0f;
#pragma unroll
  for (int j = 0; j < FDIM; ++j)
    acc += agg[n * FDIM + j] * lin[layer * FDIM * FDIM + j * FDIM + f];
  h[idx] += acc * inv_sqrt_nei;
}

// ---------------------------------------------------------------- out = segsum(h@w_out)/sqrt(6250)
#define OUT_ITER 16
__global__ __launch_bounds__(256) void k_out(const float* h, const float* w_out,
                                             const int* batch, float* out) {
  __shared__ float s_acc[GDIM * OUTD];
  for (int i = threadIdx.x; i < GDIM * OUTD; i += 256) s_acc[i] = 0.0f;
  __syncthreads();
  int base = blockIdx.x * 256 * OUT_ITER;
  for (int it = 0; it < OUT_ITER; ++it) {
    int idx = base + it * 256 + threadIdx.x;
    if (idx < N_NODES * OUTD) {
      int n = idx >> 4, f = idx & 15;
      float acc = 0.0f;
#pragma unroll
      for (int j = 0; j < FDIM; ++j) acc += h[n * FDIM + j] * w_out[j * OUTD + f];
      atomicAdd(&s_acc[batch[n] * OUTD + f], acc);   // LDS atomic (batch is sorted)
    }
  }
  __syncthreads();
  const float scale = 0.012649110640673518f;   // 1/sqrt(6250)
  for (int i = threadIdx.x; i < GDIM * OUTD; i += 256)
    atomicAdd(&out[i], s_acc[i] * scale);
}

extern "C" void kernel_launch(void* const* d_in, const int* in_sizes, int n_in,
                              void* d_out, int out_size, void* d_ws, size_t ws_size,
                              hipStream_t stream) {
  const float* pos    = (const float*)d_in[0];
  const float* x      = (const float*)d_in[1];
  const float* z      = (const float*)d_in[2];
  const float* w_in   = (const float*)d_in[3];
  const float* w_attr = (const float*)d_in[4];
  const float* fc_w1  = (const float*)d_in[5];
  const float* fc_w2  = (const float*)d_in[6];
  const float* lin    = (const float*)d_in[7];
  const float* w_out  = (const float*)d_in[8];
  const int*   esrc   = (const int*)d_in[9];
  const int*   edst   = (const int*)d_in[10];
  const int*   batch  = (const int*)d_in[11];
  float* out = (float*)d_out;

  // workspace layout
  float*  h    = (float*)d_ws;                       // N*16 f32
  float*  agg  = h + N_NODES * FDIM;                 // N*16 f32
  half_t* fc1h = (half_t*)(agg + N_NODES * FDIM);    // 2*112*32 f16
  half_t* fc2h = fc1h + 2 * H_PAD * K1_PAD;          // 2*144*128 f16

  hipMemsetAsync(out, 0, (size_t)GDIM * OUTD * sizeof(float), stream);

  int prepTot = 2 * N2 * K2_PAD;
  k_prep<<<(prepTot + 255) / 256, 256, 0, stream>>>(fc_w1, fc_w2, fc1h, fc2h);

  k_node_init<<<(N_NODES * FDIM + 255) / 256, 256, 0, stream>>>(x, z, w_in, w_attr, h);

  for (int l = 0; l < 2; ++l) {
    hipMemsetAsync(agg, 0, (size_t)N_NODES * FDIM * sizeof(float), stream);
    k_msg<<<N_EDGES / EPB, 256, 0, stream>>>(pos, h, agg, fc1h, fc2h, esrc, edst, l);
    k_update<<<(N_NODES * FDIM + 255) / 256, 256, 0, stream>>>(h, agg, lin, l);
  }

  int outBlocks = (N_NODES * OUTD + 256 * OUT_ITER - 1) / (256 * OUT_ITER);
  k_out<<<outBlocks, 256, 0, stream>>>(h, w_out, batch, out);
}
